// FeatureStereoAggrgegation_65541200937415
// MI455X (gfx1250) — matrix operand order repaired and checked by reading, not compiled, not run on hardware
//
#include <hip/hip_runtime.h>
#include <cstdint>

#define B_ 4
#define C_ 128
#define H_ 64
#define W_ 64
#define HW_ 4096
#define PC_ 128

typedef __attribute__((ext_vector_type(16))) __bf16 v16bf;
typedef __attribute__((ext_vector_type(8)))  float  v8f;
typedef __attribute__((ext_vector_type(4)))  unsigned int v4u;

union BFrag {
  v16bf v;
  v4u   q[2];
  unsigned short s[16];
};

__device__ __forceinline__ unsigned short f2bf(float f) {
  unsigned u = __float_as_uint(f);
  u += 0x7fffu + ((u >> 16) & 1u);   // round-to-nearest-even
  return (unsigned short)(u >> 16);
}

// out_cm[b][co][pix] = bias[co] + sum_ci W[co][ci] * in_cm[b][ci][pix]
__global__ __launch_bounds__(256) void proj_kernel(
    const float* __restrict__ in, const float* __restrict__ Wt,
    const float* __restrict__ bias, float* __restrict__ outp) {
  const int pix = blockIdx.x * 256 + threadIdx.x;
  const int co  = blockIdx.y * 8;
  const int bb  = blockIdx.z;
  const float* ip = in + (size_t)bb * C_ * HW_ + pix;
  float a[8];
#pragma unroll
  for (int t = 0; t < 8; ++t) a[t] = bias[co + t];
  for (int ci = 0; ci < C_; ++ci) {
    const float xv = ip[(size_t)ci * HW_];
#pragma unroll
    for (int t = 0; t < 8; ++t) a[t] += Wt[(co + t) * C_ + ci] * xv;
  }
  float* op = outp + (size_t)bb * C_ * HW_ + pix;
#pragma unroll
  for (int t = 0; t < 8; ++t) op[(size_t)(co + t) * HW_] = a[t];
}

// [B][C][HW] f32 (channel-major)  ->  [B][HW][C] bf16 (row-major), 32x32 LDS tiles
__global__ __launch_bounds__(256) void transpose_bf16_kernel(
    const float* __restrict__ in, unsigned short* __restrict__ outp) {
  __shared__ float tile[32][33];
  const int bb  = blockIdx.z;
  const int co0 = blockIdx.y * 32;
  const int px0 = blockIdx.x * 32;
  const int tx  = threadIdx.x & 31;
  const int ty  = threadIdx.x >> 5;   // 0..7
#pragma unroll
  for (int k = 0; k < 4; ++k) {
    const int r = ty + 8 * k;
    tile[r][tx] = in[(size_t)bb * C_ * HW_ + (size_t)(co0 + r) * HW_ + px0 + tx];
  }
  __syncthreads();
#pragma unroll
  for (int k = 0; k < 4; ++k) {
    const int r = ty + 8 * k;   // pixel offset in tile
    outp[(size_t)bb * HW_ * C_ + (size_t)(px0 + r) * C_ + co0 + tx] = f2bf(tile[tx][r]);
  }
}

__global__ __launch_bounds__(256) void cvt_bf16_kernel(
    const float* __restrict__ in, unsigned short* __restrict__ outp) {
  const size_t i = (size_t)blockIdx.x * 256 + threadIdx.x;
  outp[i] = f2bf(in[i]);
}

// Flash-attention (global branch), transposed formulation:
//   S^T = K @ Q^T   (A = K rows, B = Q^T)  -> keys on VGPRs, queries on lanes
//   out^T = V^T @ P^T  accumulated in f32; out^T layout == channel-major output.
// One wave owns a 16-query strip; writes out = mask * softmax(QK^T/sqrt(c)) @ V.
__global__ __launch_bounds__(256) void attn_global_kernel(
    const unsigned short* __restrict__ Qrm,   // bf16 [B][HW][C]
    const unsigned short* __restrict__ Krm,   // bf16 [B][HW][C]
    const unsigned short* __restrict__ Vt,    // bf16 [B][PC][HW]
    const float* __restrict__ mask,           // [B][HW]
    float* __restrict__ outp) {               // [B][PC][HW]
  const int wave = blockIdx.x * 8 + (threadIdx.x >> 5);
  const int lane = threadIdx.x & 31;
  const int l15  = lane & 15;
  const int half = lane >> 4;
  const int bb = wave >> 8;          // 256 strips per batch
  const int q0 = (wave & 255) << 4;  // 16 queries per wave

  constexpr float kSL2E = 0.08838834764831845f * 1.4426950408889634f; // (1/sqrt(128))*log2(e)

  // Q as B-operand: lanes 0-15 hold K(=channel) 0-7,16-23, lanes 16-31 hold 8-15,24-31 per chunk
  BFrag qB[4];
  {
    const unsigned short* qptr = Qrm + ((size_t)(bb * HW_ + q0 + l15)) * C_;
#pragma unroll
    for (int cj = 0; cj < 4; ++cj) {
      const int base = cj * 32 + half * 8;
      qB[cj].q[0] = *(const v4u*)(qptr + base);
      qB[cj].q[1] = *(const v4u*)(qptr + base + 16);
    }
  }

  v8f acc[8];   // out^T: 8 channel-tiles of 16x16
#pragma unroll
  for (int ct = 0; ct < 8; ++ct)
#pragma unroll
    for (int j = 0; j < 8; ++j) acc[ct][j] = 0.0f;

  float m_run = -1e30f;
  float l_run = 0.0f;

  for (int kv = 0; kv < HW_; kv += 32) {
    v8f st0, st1;   // S^T tiles: keys kv..kv+15 / kv+16..kv+31 (x 16 queries)
#pragma unroll
    for (int j = 0; j < 8; ++j) { st0[j] = 0.0f; st1[j] = 0.0f; }
    const unsigned short* kp0 = Krm + ((size_t)(bb * HW_ + kv + l15)) * C_;
    const unsigned short* kp1 = kp0 + 16 * C_;
#pragma unroll
    for (int cj = 0; cj < 4; ++cj) {
      // A-operand (16-bit 16x32): half0 holds K 0-7,16-23; half1 holds K 8-15,24-31
      const int lo = cj * 32 + half * 8;
      BFrag ka;
      ka.q[0] = *(const v4u*)(kp0 + lo);
      ka.q[1] = *(const v4u*)(kp0 + lo + 16);
      st0 = __builtin_amdgcn_wmma_f32_16x16x32_bf16(false, ka.v, false, qB[cj].v,
                                                    (short)0, st0, false, false);
      BFrag kb;
      kb.q[0] = *(const v4u*)(kp1 + lo);
      kb.q[1] = *(const v4u*)(kp1 + lo + 16);
      st1 = __builtin_amdgcn_wmma_f32_16x16x32_bf16(false, kb.v, false, qB[cj].v,
                                                    (short)0, st1, false, false);
    }

    // online softmax over keys (base-2 domain, scale folded in)
    float t0[8], t1[8];
    float mloc = -1e30f;
#pragma unroll
    for (int j = 0; j < 8; ++j) {
      t0[j] = st0[j] * kSL2E;
      t1[j] = st1[j] * kSL2E;
      mloc = fmaxf(mloc, fmaxf(t0[j], t1[j]));
    }
    mloc = fmaxf(mloc, __shfl_xor(mloc, 16, 32));   // combine key halves per query
    const float m_new = fmaxf(m_run, mloc);
    const float alpha = __builtin_amdgcn_exp2f(m_run - m_new);
    m_run = m_new;
    l_run *= alpha;
#pragma unroll
    for (int ct = 0; ct < 8; ++ct)
#pragma unroll
      for (int j = 0; j < 8; ++j) acc[ct][j] *= alpha;

    float p0[8], p1[8], ls = 0.0f;
#pragma unroll
    for (int j = 0; j < 8; ++j) {
      p0[j] = __builtin_amdgcn_exp2f(t0[j] - m_new);
      p1[j] = __builtin_amdgcn_exp2f(t1[j] - m_new);
      ls += p0[j] + p1[j];
    }
    l_run += ls;

    // build P^T as B-operand: lanes 0-15 need keys 0..15, lanes 16-31 keys 16..31.
    // S^T tile layout puts keys 8-15 / 16-23 on the partner half -> one xor-16 exchange.
    BFrag pB;
#pragma unroll
    for (int j = 0; j < 8; ++j) {
      const float po0 = __shfl_xor(p0[j], 16, 32);
      const float po1 = __shfl_xor(p1[j], 16, 32);
      pB.s[j]     = f2bf(half ? po1 : p0[j]);
      pB.s[8 + j] = f2bf(half ? p1[j] : po0);
    }

    // out^T += V^T(tile) @ P^T ; V^T rows are contiguous keys (channel-major V)
    const unsigned short* vb = Vt + ((size_t)(bb * PC_ + l15)) * HW_ + kv + half * 16;
#pragma unroll
    for (int ct = 0; ct < 8; ++ct) {
      BFrag va;
      const unsigned short* vp = vb + (size_t)(ct * 16) * HW_;
      va.q[0] = *(const v4u*)(vp);
      va.q[1] = *(const v4u*)(vp + 8);
      acc[ct] = __builtin_amdgcn_wmma_f32_16x16x32_bf16(false, va.v, false, pB.v,
                                                        (short)0, acc[ct], false, false);
    }
  }

  const float lt = l_run + __shfl_xor(l_run, 16, 32);
  const float mv = mask[bb * HW_ + q0 + l15];
  const float factor = mv / lt;    // mask * softmax-normalization
#pragma unroll
  for (int ct = 0; ct < 8; ++ct) {
#pragma unroll
    for (int vi = 0; vi < 8; ++vi) {
      const int ch = ct * 16 + vi + 8 * half;
      outp[((size_t)(bb * PC_ + ch)) * HW_ + q0 + l15] = acc[ct][vi] * factor;
    }
  }
}

// Local 3x3 window branch; adds (1-mask)*out_local into d_out (global wrote first).
__global__ __launch_bounds__(256) void attn_local_kernel(
    const float* __restrict__ qcm, const float* __restrict__ kcm,
    const float* __restrict__ vv, const float* __restrict__ mask,
    float* __restrict__ outp) {
  const int id  = blockIdx.x * 256 + threadIdx.x;   // 0 .. B*HW-1
  const int bb  = id >> 12;
  const int pix = id & (HW_ - 1);
  const int y = pix >> 6, x = pix & 63;

  constexpr float kSL2E = 0.08838834764831845f * 1.4426950408889634f;

  int   npix[9];
  float vfac[9];
#pragma unroll
  for (int dy = 0; dy < 3; ++dy)
#pragma unroll
    for (int dx = 0; dx < 3; ++dx) {
      const int j = dy * 3 + dx;
      const int py = y + dy - 1, px = x + dx - 1;
      const bool ok = (py >= 0) & (py < H_) & (px >= 0) & (px < W_);
      npix[j] = ok ? (py * W_ + px) : pix;
      vfac[j] = ok ? 1.0f : 0.0f;   // zero-padded k/v windows
    }

  const float* qb = qcm + (size_t)bb * C_ * HW_ + pix;
  const float* kb = kcm + (size_t)bb * C_ * HW_;
  float s[9];
#pragma unroll
  for (int j = 0; j < 9; ++j) s[j] = 0.0f;
  for (int ci = 0; ci < C_; ++ci) {
    const float qv = qb[(size_t)ci * HW_];
    const float* kr = kb + (size_t)ci * HW_;
#pragma unroll
    for (int j = 0; j < 9; ++j) s[j] += qv * kr[npix[j]] * vfac[j];
  }
  float mx = -1e30f;
#pragma unroll
  for (int j = 0; j < 9; ++j) { s[j] *= kSL2E; mx = fmaxf(mx, s[j]); }
  float p[9], sum = 0.0f;
#pragma unroll
  for (int j = 0; j < 9; ++j) { p[j] = __builtin_amdgcn_exp2f(s[j] - mx); sum += p[j]; }
  const float inv = 1.0f / sum;
  float wgt[9];
#pragma unroll
  for (int j = 0; j < 9; ++j) wgt[j] = p[j] * inv * vfac[j];

  const float om = 1.0f - mask[bb * HW_ + pix];
  const float* vb = vv + (size_t)bb * PC_ * HW_;
  float* ob = outp + (size_t)bb * PC_ * HW_ + pix;
  for (int ci = 0; ci < PC_; ++ci) {
    const float* vr = vb + (size_t)ci * HW_;
    float a = 0.0f;
#pragma unroll
    for (int j = 0; j < 9; ++j) a += wgt[j] * vr[npix[j]];
    ob[(size_t)ci * HW_] += om * a;
  }
}

extern "C" void kernel_launch(void* const* d_in, const int* in_sizes, int n_in,
                              void* d_out, int out_size, void* d_ws, size_t ws_size,
                              hipStream_t stream) {
  const float* x    = (const float*)d_in[0];   // [B][C][H][W]
  const float* msk  = (const float*)d_in[1];   // [B][1][H][W]
  const float* aggr = (const float*)d_in[2];   // [B][PC][H][W]
  const float* Wq   = (const float*)d_in[3];
  const float* bq   = (const float*)d_in[4];
  const float* Wk   = (const float*)d_in[5];
  const float* bk   = (const float*)d_in[6];
  float* out = (float*)d_out;

  // workspace carve (all fully overwritten before use)
  const size_t SZ_CM = (size_t)B_ * C_ * HW_ * sizeof(float);          // 8 MB
  const size_t SZ_BF = (size_t)B_ * C_ * HW_ * sizeof(unsigned short); // 4 MB
  char* ws = (char*)d_ws;
  float* qcm  = (float*)ws;            ws += SZ_CM;
  float* kimg = (float*)ws;            ws += SZ_CM;
  float* kglb = (float*)ws;            ws += SZ_CM;
  unsigned short* qrm = (unsigned short*)ws; ws += SZ_BF;
  unsigned short* krm = (unsigned short*)ws; ws += SZ_BF;
  unsigned short* vt  = (unsigned short*)ws; ws += SZ_BF;

  const dim3 projGrid(HW_ / 256, C_ / 8, B_);
  proj_kernel<<<projGrid, 256, 0, stream>>>(x,   Wq, bq, qcm);   // q (both branches)
  proj_kernel<<<projGrid, 256, 0, stream>>>(x,   Wk, bk, kimg);  // k_proj(context), local
  proj_kernel<<<projGrid, 256, 0, stream>>>(qcm, Wk, bk, kglb);  // k_proj(q), global

  const dim3 trGrid(HW_ / 32, C_ / 32, B_);
  transpose_bf16_kernel<<<trGrid, 256, 0, stream>>>(qcm,  qrm);
  transpose_bf16_kernel<<<trGrid, 256, 0, stream>>>(kglb, krm);
  cvt_bf16_kernel<<<(B_ * PC_ * HW_) / 256, 256, 0, stream>>>(aggr, vt);

  // 1024 waves: 4 batches x 256 query strips; 8 waves per block
  attn_global_kernel<<<(B_ * (HW_ / 16)) / 8, 256, 0, stream>>>(qrm, krm, vt, msk, out);
  attn_local_kernel<<<(B_ * HW_) / 256, 256, 0, stream>>>(qcm, kimg, aggr, msk, out);
}
